// SIM_Mono_50328426774593
// MI455X (gfx1250) — compile-verified, matrix operand order checked by reading, not run on hardware
//
#include <hip/hip_runtime.h>
#include <hip/hip_bf16.h>

typedef __attribute__((ext_vector_type(16))) __bf16          v16bf;
typedef __attribute__((ext_vector_type(8)))  float           v8f;
typedef __attribute__((ext_vector_type(8)))  unsigned short  u16x8;
typedef __attribute__((ext_vector_type(16))) unsigned short  u16x16;
typedef __attribute__((ext_vector_type(4)))  float           f32x4;
typedef unsigned short u16;

#define C_DIM 256
#define W_DIM 128
#define H_DIM 128
#define HW    (H_DIM * W_DIM)

__device__ __forceinline__ u16 f2bf(float f) {
  unsigned int u = __builtin_bit_cast(unsigned int, f);
  u += 0x7FFFu + ((u >> 16) & 1u);            // round-to-nearest-even
  return (u16)(u >> 16);
}
__device__ __forceinline__ float bf2f(u16 h) {
  unsigned int u = ((unsigned int)h) << 16;
  return __builtin_bit_cast(float, u);
}

// 16 bf16 halves per lane: two contiguous 8-half runs at +0 and +16 halves
// (16-bit A/B operand K-layout: K = (i&7) + ((i>>3)<<4) [+8 for hi lanes])
__device__ __forceinline__ v16bf load_bf_frag(const u16* p) {
  u16x8 lo = *(const u16x8*)(p);
  u16x8 hi = *(const u16x8*)(p + 16);
  u16x16 v;
#pragma unroll
  for (int i = 0; i < 8; ++i) { v[i] = lo[i]; v[8 + i] = hi[i]; }
  return __builtin_bit_cast(v16bf, v);
}

// raw 16-float gather (two 8-float runs) for the attention-prob A operand
struct Raw16 { f32x4 r0, r1, r2, r3; };
__device__ __forceinline__ Raw16 load_raw16(const float* p) {
  Raw16 r;
  r.r0 = *(const f32x4*)(p);
  r.r1 = *(const f32x4*)(p + 4);
  r.r2 = *(const f32x4*)(p + 16);
  r.r3 = *(const f32x4*)(p + 20);
  return r;
}
__device__ __forceinline__ v16bf cvt_raw16(const Raw16& r) {
  u16x16 v;
#pragma unroll
  for (int i = 0; i < 4; ++i) {
    v[i]      = f2bf(r.r0[i]);
    v[4 + i]  = f2bf(r.r1[i]);
    v[8 + i]  = f2bf(r.r2[i]);
    v[12 + i] = f2bf(r.r3[i]);
  }
  return __builtin_bit_cast(v16bf, v);
}

__device__ __forceinline__ v8f wmma_bf16(v16bf a, v16bf b, v8f c) {
  return __builtin_amdgcn_wmma_f32_16x16x32_bf16(false, a, false, b, (short)0, c,
                                                 false, false);
}

// ---------------------------------------------------------------------------
// Pre-pass 1: fused-LayerNorm affine terms.
//   rowsum_*[o] = sum_c W[o][c]*norm_w[c];  bias_*[o] = b[o] + sum_c W[o][c]*norm_b[c]
// ---------------------------------------------------------------------------
__global__ void prep_affine(const float* __restrict__ wq, const float* __restrict__ bq,
                            const float* __restrict__ wk, const float* __restrict__ bk,
                            const float* __restrict__ norm_w, const float* __restrict__ norm_b,
                            float* __restrict__ rowsum_q, float* __restrict__ bias_q,
                            float* __restrict__ rowsum_k, float* __restrict__ bias_k) {
  int o = threadIdx.x;
  const float* Wm = (blockIdx.x == 0) ? wq : wk;
  const float* bb = (blockIdx.x == 0) ? bq : bk;
  float rs = 0.f, bn = 0.f;
  for (int c = 0; c < C_DIM; ++c) {
    float v = Wm[o * C_DIM + c];
    rs += v * norm_w[c];
    bn += v * norm_b[c];
  }
  if (blockIdx.x == 0) { rowsum_q[o] = rs; bias_q[o] = bb[o] + bn; }
  else                 { rowsum_k[o] = rs; bias_k[o] = bb[o] + bn; }
}

// ---------------------------------------------------------------------------
// Pre-pass 2: weights -> bf16, WMMA A-operand fragment order
//   layout: [(tm*8 + kc)*32 + lane] * 16 halves (contiguous 32B per lane)
//   wq,wk get norm_w folded in (Wq'[o][c] = wq[o][c]*norm_w[c])
// ---------------------------------------------------------------------------
__global__ void prep_wfrag(const float* __restrict__ wq, const float* __restrict__ wk,
                           const float* __restrict__ wv, const float* __restrict__ norm_w,
                           u16* __restrict__ fq, u16* __restrict__ fk, u16* __restrict__ fv) {
  int gid = blockIdx.x * 256 + threadIdx.x;       // 0 .. 3*4096-1
  int s = gid / 4096;
  int r = gid % 4096;
  int tm   = r >> 8;
  int kc   = (r >> 5) & 7;
  int lane = r & 31;
  const float* Wm = (s == 0) ? wq : (s == 1) ? wk : wv;
  u16* dst = ((s == 0) ? fq : (s == 1) ? fk : fv) + (size_t)r * 16;
  int m  = tm * 16 + (lane & 15);
  int hi = (lane >> 4) << 3;
#pragma unroll
  for (int i = 0; i < 16; ++i) {
    int k = kc * 32 + (i & 7) + ((i >> 3) << 4) + hi;
    float v = Wm[m * C_DIM + k];
    if (s < 2) v *= norm_w[k];
    dst[i] = f2bf(v);
  }
}

// ---------------------------------------------------------------------------
// Main fused kernel: one workgroup per (b,h) row, 8 waves, 256 threads.
// Stage 2: waves partition OUTPUT-CHANNEL tiles -> weights loaded exactly once
//   per workgroup; each LDS X-fragment feeds two WMMAs; streamed operand is
//   software-pipelined one step ahead to hide LDS latency.
// Stages 3/5: waves own fixed N-tiles -> B operands hoisted into registers,
//   streamed A operand pipelined.
// LDS (265 KB dynamic):
//   [0,64K)    XT bf16 [w][c]  -> reused as S f32 [w][v]
//   [64,128K)  QT bf16 [w][c]  -> reused as FT bf16 [w][c]
//   [128,192K) KT bf16 [w][c]
//   [192,256K) VB bf16 [c][v]
//   [256K,..)  stats: mu[128], rinv[128], red[512]
// ---------------------------------------------------------------------------
__global__ __launch_bounds__(256, 1)
void axial_attn_kernel(const float* __restrict__ x,
                       const u16* __restrict__ fq, const u16* __restrict__ fk,
                       const u16* __restrict__ fv,
                       const float* __restrict__ rowsum_q, const float* __restrict__ bias_q,
                       const float* __restrict__ rowsum_k, const float* __restrict__ bias_k,
                       const float* __restrict__ bv, const float* __restrict__ beta,
                       float* __restrict__ out) {
  extern __shared__ char smem[];
  u16*   XT    = (u16*)smem;
  float* S     = (float*)smem;
  u16*   QT    = (u16*)(smem + 65536);
  u16*   FT    = QT;
  u16*   KT    = (u16*)(smem + 131072);
  u16*   VB    = (u16*)(smem + 196608);
  float* stats = (float*)(smem + 262144);
  float* mu    = stats;
  float* rinv  = stats + 128;
  float* red   = stats + 256;

  const int t    = threadIdx.x;
  const int lane = t & 31;
  const int wave = t >> 5;
  const int bh   = blockIdx.x;
  const int bidx = bh >> 7;
  const int hidx = bh & 127;
  const float* xrow = x + (size_t)bidx * C_DIM * HW + (size_t)hidx * W_DIM;

  const int hioff = (lane >> 4) << 3;   // +8: K-offset for hi lanes AND row offset of C/D
  const int nsub  = lane & 15;

  // ---- Stage 1: stage X row as bf16 (transposed) + LayerNorm stats ----
  {
    int w = t & 127, half = t >> 7;
    float s1 = 0.f, s2 = 0.f;
    int c0 = half * 128;
#pragma unroll 4
    for (int c = c0; c < c0 + 128; ++c) {
      float v = xrow[(size_t)c * HW + w];
      s1 += v; s2 += v * v;
      XT[w * 256 + c] = f2bf(v);
    }
    red[t] = s1; red[256 + t] = s2;
  }
  __syncthreads();
  if (t < 128) {
    float m = (red[t] + red[t + 128]) * (1.0f / 256.0f);
    float q = (red[256 + t] + red[256 + t + 128]) * (1.0f / 256.0f);
    mu[t]   = m;
    rinv[t] = rsqrtf(q - m * m + 1e-6f);
  }
  __syncthreads();

  // ---- Stage 2: Q/K/V projections ----
  {
    const int tm0 = wave * 2, tm1 = tm0 + 1;

    for (int s = 0; s < 3; ++s) {
      const u16* WF = (s == 0) ? fq : (s == 1) ? fk : fv;

      v16bf a0[8], a1[8];
#pragma unroll
      for (int kc = 0; kc < 8; ++kc) {
        a0[kc] = __builtin_bit_cast(
            v16bf, *(const u16x16*)(WF + (size_t)((tm0 * 8 + kc) * 32 + lane) * 16));
        a1[kc] = __builtin_bit_cast(
            v16bf, *(const u16x16*)(WF + (size_t)((tm1 * 8 + kc) * 32 + lane) * 16));
      }

      // per-channel affine terms for this wave's output rows (registers)
      float rs0[8], bi0[8], rs1[8], bi1[8];
      if (s < 2) {
        const float* rsum = (s == 0) ? rowsum_q : rowsum_k;
        const float* bia  = (s == 0) ? bias_q  : bias_k;
#pragma unroll
        for (int r = 0; r < 8; ++r) {
          rs0[r] = rsum[tm0 * 16 + r + hioff]; bi0[r] = bia[tm0 * 16 + r + hioff];
          rs1[r] = rsum[tm1 * 16 + r + hioff]; bi1[r] = bia[tm1 * 16 + r + hioff];
        }
      } else {
#pragma unroll
        for (int r = 0; r < 8; ++r) {
          bi0[r] = bv[tm0 * 16 + r + hioff];
          bi1[r] = bv[tm1 * 16 + r + hioff];
          rs0[r] = 0.f; rs1[r] = 0.f;
        }
      }

      for (int tn = 0; tn < 8; ++tn) {
        const u16* bbase = XT + (tn * 16 + nsub) * 256 + hioff;
        v8f acc0 = {}, acc1 = {};
        v16bf bcur = load_bf_frag(bbase);
#pragma unroll
        for (int kc = 0; kc < 8; ++kc) {
          v16bf bnext = bcur;
          if (kc < 7) bnext = load_bf_frag(bbase + (kc + 1) * 32);  // prefetch
          acc0 = wmma_bf16(a0[kc], bcur, acc0);
          acc1 = wmma_bf16(a1[kc], bcur, acc1);
          bcur = bnext;
        }
        int wcol = tn * 16 + nsub;
        if (s == 2) {
#pragma unroll
          for (int r = 0; r < 8; ++r) {
            VB[(tm0 * 16 + r + hioff) * 128 + wcol] = f2bf(acc0[r] + bi0[r]);
            VB[(tm1 * 16 + r + hioff) * 128 + wcol] = f2bf(acc1[r] + bi1[r]);
          }
        } else {
          u16* dstq = (s == 0) ? QT : KT;
          float r_w = rinv[wcol], m_w = mu[wcol];   // 2x ds_load_b32, no movrel
#pragma unroll
          for (int r = 0; r < 8; ++r) {
            dstq[wcol * 256 + tm0 * 16 + r + hioff] =
                f2bf(r_w * acc0[r] - r_w * m_w * rs0[r] + bi0[r]);
            dstq[wcol * 256 + tm1 * 16 + r + hioff] =
                f2bf(r_w * acc1[r] - r_w * m_w * rs1[r] + bi1[r]);
          }
        }
      }
    }
  }
  __syncthreads();            // QT/KT/VB ready; XT dead

  // ---- Stage 3: S = (Q^T K) * C^-1/2. K frags hoisted; Q frags pipelined ----
  {
    v16bf kb[8];
    const u16* kbase = KT + (wave * 16 + nsub) * 256 + hioff;
#pragma unroll
    for (int kc = 0; kc < 8; ++kc) kb[kc] = load_bf_frag(kbase + kc * 32);

    const float scale = 0.0625f;   // 256^-0.5
    for (int tm = 0; tm < 8; ++tm) {
      const u16* qbase = QT + (tm * 16 + nsub) * 256 + hioff;
      v8f acc = {};
      v16bf acur = load_bf_frag(qbase);
#pragma unroll
      for (int kc = 0; kc < 8; ++kc) {
        v16bf anext = acur;
        if (kc < 7) anext = load_bf_frag(qbase + (kc + 1) * 32);   // prefetch
        acc = wmma_bf16(acur, kb[kc], acc);
        acur = anext;
      }
#pragma unroll
      for (int r = 0; r < 8; ++r)
        S[(tm * 16 + r + hioff) * 128 + wave * 16 + nsub] = acc[r] * scale;
    }
  }
  __syncthreads();

  // ---- Stage 4: row softmax (f32, in LDS) ----
  {
    int row = t & 127, half = t >> 7;
    int v0 = half * 64;
    float mx = -3.4e38f;
    for (int v = v0; v < v0 + 64; ++v) mx = fmaxf(mx, S[row * 128 + v]);
    red[t] = mx;
    __syncthreads();
    float rmax = fmaxf(red[row], red[row + 128]);
    float sum = 0.f;
    for (int v = v0; v < v0 + 64; ++v) {
      float e = __expf(S[row * 128 + v] - rmax);
      S[row * 128 + v] = e;
      sum += e;
    }
    red[256 + t] = sum;
    __syncthreads();
    float inv = 1.0f / (red[256 + row] + red[256 + row + 128]);
    for (int v = v0; v < v0 + 64; ++v) S[row * 128 + v] *= inv;
  }
  __syncthreads();

  // ---- Stage 5: F^T = P * V^T. V frags hoisted; raw prob loads pipelined,
  //      one converted A frag feeds both accumulators ----
  {
    v16bf vb0[4], vb1[4];
#pragma unroll
    for (int kc = 0; kc < 4; ++kc) {
      vb0[kc] = load_bf_frag(VB + ((wave    ) * 16 + nsub) * 128 + kc * 32 + hioff);
      vb1[kc] = load_bf_frag(VB + ((wave + 8) * 16 + nsub) * 128 + kc * 32 + hioff);
    }
    for (int tm = 0; tm < 8; ++tm) {
      const float* pbase = S + (tm * 16 + nsub) * 128 + hioff;
      v8f acc0 = {}, acc1 = {};
      Raw16 rcur = load_raw16(pbase);
#pragma unroll
      for (int kc = 0; kc < 4; ++kc) {
        Raw16 rnext = rcur;
        if (kc < 3) rnext = load_raw16(pbase + (kc + 1) * 32);     // prefetch
        v16bf a = cvt_raw16(rcur);
        acc0 = wmma_bf16(a, vb0[kc], acc0);
        acc1 = wmma_bf16(a, vb1[kc], acc1);
        rcur = rnext;
      }
#pragma unroll
      for (int r = 0; r < 8; ++r) {
        int wrow = (tm * 16 + r + hioff) * 256;
        FT[wrow + (wave    ) * 16 + nsub] = f2bf(acc0[r]);
        FT[wrow + (wave + 8) * 16 + nsub] = f2bf(acc1[r]);
      }
    }
  }
  __syncthreads();

  // ---- Stage 6: out = x + beta[c] * F  (coalesced global pass) ----
  for (int i = 0; i < 128; ++i) {
    int idx = i * 256 + t;
    int c = idx >> 7, w = idx & 127;
    float f = bf2f(FT[w * 256 + c]);
    size_t g = (size_t)c * HW + w;
    out[(size_t)bidx * C_DIM * HW + (size_t)hidx * W_DIM + g] = xrow[g] + f * beta[c];
  }
}

// ---------------------------------------------------------------------------
// Host entry. Workspace layout (needs ~388 KB):
//   [0,128K) fq | [128K,256K) fk | [256K,384K) fv | then 4x 1KB f32 arrays
// ---------------------------------------------------------------------------
extern "C" void kernel_launch(void* const* d_in, const int* in_sizes, int n_in,
                              void* d_out, int out_size, void* d_ws, size_t ws_size,
                              hipStream_t stream) {
  (void)in_sizes; (void)n_in; (void)out_size; (void)ws_size;
  const float* x      = (const float*)d_in[0];
  const float* norm_w = (const float*)d_in[1];
  const float* norm_b = (const float*)d_in[2];
  const float* wq     = (const float*)d_in[3];
  const float* bq     = (const float*)d_in[4];
  const float* wk     = (const float*)d_in[5];
  const float* bk     = (const float*)d_in[6];
  const float* wv     = (const float*)d_in[7];
  const float* bv     = (const float*)d_in[8];
  const float* beta   = (const float*)d_in[9];
  float* out = (float*)d_out;

  char* ws = (char*)d_ws;
  u16*   fq       = (u16*)(ws);
  u16*   fk       = (u16*)(ws + 131072);
  u16*   fv       = (u16*)(ws + 262144);
  float* rowsum_q = (float*)(ws + 393216);
  float* rowsum_k = (float*)(ws + 394240);
  float* bias_q   = (float*)(ws + 395264);
  float* bias_k   = (float*)(ws + 396288);

  prep_affine<<<dim3(2), dim3(256), 0, stream>>>(wq, bq, wk, bk, norm_w, norm_b,
                                                 rowsum_q, bias_q, rowsum_k, bias_k);
  prep_wfrag<<<dim3(48), dim3(256), 0, stream>>>(wq, wk, wv, norm_w, fq, fk, fv);

  const size_t smem_bytes = 262144 + 3072;   // 265 KB < 320 KB WGP LDS
  axial_attn_kernel<<<dim3(1024), dim3(256), smem_bytes, stream>>>(
      x, fq, fk, fv, rowsum_q, bias_q, rowsum_k, bias_k, bv, beta, out);
}
